// Model2_4492535791937
// MI455X (gfx1250) — matrix operand order repaired and checked by reading, not compiled, not run on hardware
//
#include <hip/hip_runtime.h>

typedef __attribute__((ext_vector_type(16))) __bf16 v16bf;
typedef __attribute__((ext_vector_type(8)))  float  v8f;

#define B_   64
#define T_   1024
#define BT   65536
#define IN_  512
#define H1_  64
#define H2_  6

__device__ __forceinline__ float sigf(float x) { return 1.f / (1.f + __expf(-x)); }

// -----------------------------------------------------------------------------
// Kernel 1: G[m,n] = sum_k X[m,k]*W[n,k] + bias[n], N fixed = 256 (4*H1).
// bf16 WMMA, f32 accumulate. WG tile 128(M) x 256(N): 8 waves = 4 Mquads x 2 Nhalves,
// each wave 2 Mtiles x 8 Ntiles (16 accumulators). Software-pipelined LDS staging:
// regs->LDS, barrier, issue next K-step global loads, then 16 WMMAs.
// -----------------------------------------------------------------------------
__global__ __launch_bounds__(256) void gates256_wmma(
    const float* __restrict__ X, const float* __restrict__ W,
    const float* __restrict__ bias, float* __restrict__ G, int K)
{
    __shared__ __bf16 As[128][40];   // [m][k], padded
    __shared__ __bf16 Bs[256][40];   // [n][k], padded (transposed staging)

    const int tid  = threadIdx.x;
    const int lane = tid & 31;
    const int wv   = tid >> 5;
    const int mq   = wv >> 1;        // 0..3 -> rows [mq*32, mq*32+32)
    const int nh   = wv & 1;         // 0..1 -> cols [nh*128, nh*128+128)
    const int m0   = blockIdx.x * 128;

    const int ar = tid >> 1;          // A stage: row 0..127
    const int akq = (tid & 1) * 16;   //          k offset 0 or 16
    const float* Arow = X + (size_t)(m0 + ar) * K + akq;
    const float* Brow = W + (size_t)tid * K;   // B stage: n = tid

    float4 av[4];   // 16 staged A values
    float4 bv[8];   // 32 staged B values

    v8f zero = {};
    v8f acc[2][8];
    #pragma unroll
    for (int i = 0; i < 2; ++i)
        #pragma unroll
        for (int nt = 0; nt < 8; ++nt) acc[i][nt] = zero;

    // prologue: stage K-step 0 into registers
    #pragma unroll
    for (int q = 0; q < 4; ++q) av[q] = *(const float4*)(Arow + 4 * q);
    #pragma unroll
    for (int q = 0; q < 8; ++q) bv[q] = *(const float4*)(Brow + 4 * q);

    for (int k0 = 0; k0 < K; k0 += 32) {
        // regs -> LDS (bf16 convert)
        #pragma unroll
        for (int q = 0; q < 4; ++q) {
            As[ar][akq + 4 * q + 0] = (__bf16)av[q].x;
            As[ar][akq + 4 * q + 1] = (__bf16)av[q].y;
            As[ar][akq + 4 * q + 2] = (__bf16)av[q].z;
            As[ar][akq + 4 * q + 3] = (__bf16)av[q].w;
        }
        #pragma unroll
        for (int q = 0; q < 8; ++q) {
            Bs[tid][4 * q + 0] = (__bf16)bv[q].x;
            Bs[tid][4 * q + 1] = (__bf16)bv[q].y;
            Bs[tid][4 * q + 2] = (__bf16)bv[q].z;
            Bs[tid][4 * q + 3] = (__bf16)bv[q].w;
        }
        __syncthreads();

        // issue next K-step global loads; they overlap the WMMAs below
        if (k0 + 32 < K) {
            #pragma unroll
            for (int q = 0; q < 4; ++q) av[q] = *(const float4*)(Arow + (k0 + 32) + 4 * q);
            #pragma unroll
            for (int q = 0; q < 8; ++q) bv[q] = *(const float4*)(Brow + (k0 + 32) + 4 * q);
        }

        // A fragments (ISA 16x32 bf16 layout: lane = M row, K split by lane half)
        v16bf a[2];
        #pragma unroll
        for (int i = 0; i < 2; ++i) {
            int m    = mq * 32 + i * 16 + (lane & 15);
            int koff = (lane >> 4) * 8;
            #pragma unroll
            for (int e = 0; e < 16; ++e)
                a[i][e] = As[m][koff + e + ((e >= 8) ? 8 : 0)];
        }
        #pragma unroll
        for (int nt = 0; nt < 8; ++nt) {
            v16bf b;
            int n    = nh * 128 + nt * 16 + (lane & 15);
            int koff = (lane >> 4) * 8;
            #pragma unroll
            for (int e = 0; e < 16; ++e) b[e] = Bs[n][koff + e + ((e >= 8) ? 8 : 0)];
            acc[0][nt] = __builtin_amdgcn_wmma_f32_16x16x32_bf16(
                false, a[0], false, b, (short)0, acc[0][nt], false, false);
            acc[1][nt] = __builtin_amdgcn_wmma_f32_16x16x32_bf16(
                false, a[1], false, b, (short)0, acc[1][nt], false, false);
        }
        __syncthreads();
    }

    #pragma unroll
    for (int nt = 0; nt < 8; ++nt) {
        int   n  = nh * 128 + nt * 16 + (lane & 15);
        float bvv = bias[n];
        #pragma unroll
        for (int i = 0; i < 2; ++i) {
            #pragma unroll
            for (int v = 0; v < 8; ++v) {
                int m = m0 + mq * 32 + i * 16 + v + ((lane >> 4) ? 8 : 0);
                G[(size_t)m * 256 + n] = acc[i][nt][v] + bvv;
            }
        }
    }
}

// -----------------------------------------------------------------------------
// Kernel 2: persistent LSTM recurrence, H=64. 4 WGs x 16 batch rows x 128 threads.
// Wave j owns n-tiles {j, j+4, j+8, j+12} => it holds i,f,g,o for the SAME (m,h)
// positions in matching accumulator lanes: gate update stays in registers, no
// gate LDS exchange. Only the 2KB bf16 h-state crosses LDS. xg[t+1] loads are
// pipelined behind step t's WMMA + VALU.
// -----------------------------------------------------------------------------
__global__ __launch_bounds__(128) void lstm64_wmma(
    const float* __restrict__ xg,   // [BT, 256], row = b*T + t
    const float* __restrict__ Whh,  // [256, 64]
    float* __restrict__ hout)       // [BT, 64]
{
    __shared__ __bf16 Hs[16][72];   // h state, bf16, padded

    const int tid  = threadIdx.x;
    const int lane = tid & 31;
    const int wv   = tid >> 5;               // 0..3
    const int b0   = blockIdx.x * 16;
    const int hcol = wv * 16 + (lane & 15);  // this lane's h column
    const int mhi  = (lane >> 4) ? 8 : 0;    // +8 row offset for upper half-wave

    // Preload Whh B-fragments: [gate][kstep]; gate g -> n-tile g*4 + wv
    v16bf bw[4][2];
    #pragma unroll
    for (int g = 0; g < 4; ++g) {
        int n = g * 64 + hcol;
        #pragma unroll
        for (int ks = 0; ks < 2; ++ks) {
            int kbase = ks * 32 + (lane >> 4) * 8;
            #pragma unroll
            for (int e = 0; e < 16; ++e)
                bw[g][ks][e] = (__bf16)Whh[(size_t)n * 64 + kbase + e + ((e >= 8) ? 8 : 0)];
        }
    }

    for (int i = tid; i < 16 * 72; i += 128) ((__bf16*)Hs)[i] = (__bf16)0.f;

    float cst[8];
    #pragma unroll
    for (int v = 0; v < 8; ++v) cst[v] = 0.f;

    // pipelined xg staging: xv = step t, xn = step t+1
    float xv[32], xn[32];
    #pragma unroll
    for (int v = 0; v < 8; ++v) {
        const float* xr = xg + ((size_t)(b0 + v + mhi) * T_ + 0) * 256 + hcol;
        xv[0 * 8 + v] = xr[0];   xv[1 * 8 + v] = xr[64];
        xv[2 * 8 + v] = xr[128]; xv[3 * 8 + v] = xr[192];
    }
    __syncthreads();

    for (int t = 0; t < T_; ++t) {
        // A fragments from h state (K = 0..31 and 32..63)
        v16bf a0, a1;
        {
            int m    = lane & 15;
            int koff = (lane >> 4) * 8;
            #pragma unroll
            for (int e = 0; e < 16; ++e) {
                int kl = koff + e + ((e >= 8) ? 8 : 0);
                a0[e] = Hs[m][kl];
                a1[e] = Hs[m][32 + kl];
            }
        }
        __syncthreads();   // all reads of Hs done before anyone rewrites it

        v8f zero = {};
        v8f ga[4];
        #pragma unroll
        for (int g = 0; g < 4; ++g) {
            v8f acc = zero;
            acc = __builtin_amdgcn_wmma_f32_16x16x32_bf16(false, a0, false, bw[g][0], (short)0, acc, false, false);
            acc = __builtin_amdgcn_wmma_f32_16x16x32_bf16(false, a1, false, bw[g][1], (short)0, acc, false, false);
            ga[g] = acc;
        }

        // prefetch next timestep's xg while the update math runs
        if (t + 1 < T_) {
            #pragma unroll
            for (int v = 0; v < 8; ++v) {
                const float* xr = xg + ((size_t)(b0 + v + mhi) * T_ + (t + 1)) * 256 + hcol;
                xn[0 * 8 + v] = xr[0];   xn[1 * 8 + v] = xr[64];
                xn[2 * 8 + v] = xr[128]; xn[3 * 8 + v] = xr[192];
            }
        }

        // in-register gate update (pytorch order i,f,g,o)
        #pragma unroll
        for (int v = 0; v < 8; ++v) {
            float ig = sigf (ga[0][v] + xv[0 * 8 + v]);
            float fg = sigf (ga[1][v] + xv[1 * 8 + v]);
            float gg = tanhf(ga[2][v] + xv[2 * 8 + v]);
            float og = sigf (ga[3][v] + xv[3 * 8 + v]);
            float c  = fg * cst[v] + ig * gg;
            cst[v]   = c;
            float h  = og * tanhf(c);
            int m    = v + mhi;
            Hs[m][hcol] = (__bf16)h;
            hout[((size_t)(b0 + m) * T_ + t) * 64 + hcol] = h;
        }
        #pragma unroll
        for (int i = 0; i < 32; ++i) xv[i] = xn[i];
        __syncthreads();   // Hs fully updated before next step's fragment reads
    }
}

// -----------------------------------------------------------------------------
// Kernel 3: small gate GEMM, N=24, K in {64, 6}. Plain f32; W cached in LDS.
// -----------------------------------------------------------------------------
__global__ __launch_bounds__(256) void gates_small(
    const float* __restrict__ X, const float* __restrict__ W,
    const float* __restrict__ bias, float* __restrict__ G, int K)
{
    __shared__ float Ws[24 * 64];
    for (int i = threadIdx.x; i < 24 * K; i += 256) Ws[i] = W[i];
    __syncthreads();
    size_t idx = (size_t)blockIdx.x * 256 + threadIdx.x;
    if (idx >= (size_t)BT * 24) return;
    int m = (int)(idx / 24), n = (int)(idx % 24);
    const float* xr = X + (size_t)m * K;
    const float* wr = Ws + n * K;
    float s = bias[n];
    for (int k = 0; k < K; ++k) s += xr[k] * wr[k];
    G[idx] = s;
}

// -----------------------------------------------------------------------------
// Kernel 4: LSTM recurrence, H=6. One thread per batch row, state in registers.
// -----------------------------------------------------------------------------
__global__ __launch_bounds__(64) void lstm6(
    const float* __restrict__ xg,   // [BT, 24]
    const float* __restrict__ Whh,  // [24, 6]
    float* __restrict__ hout)       // [BT, 6]
{
    __shared__ float Ws[24 * 6];
    for (int i = threadIdx.x; i < 144; i += 64) Ws[i] = Whh[i];
    __syncthreads();
    const int b = threadIdx.x;
    float h[6] = {0, 0, 0, 0, 0, 0};
    float c[6] = {0, 0, 0, 0, 0, 0};
    const float* xr = xg + (size_t)b * T_ * 24;
    float*       hr = hout + (size_t)b * T_ * 6;
    for (int t = 0; t < T_; ++t) {
        float g[24];
        #pragma unroll
        for (int n = 0; n < 24; ++n) {
            float s = xr[t * 24 + n];
            #pragma unroll
            for (int k = 0; k < 6; ++k) s += h[k] * Ws[n * 6 + k];
            g[n] = s;
        }
        #pragma unroll
        for (int j = 0; j < 6; ++j) {
            float ig = sigf(g[j]);
            float fg = sigf(g[6 + j]);
            float gg = tanhf(g[12 + j]);
            float og = sigf(g[18 + j]);
            c[j] = fg * c[j] + ig * gg;
            h[j] = og * tanhf(c[j]);
            hr[t * 6 + j] = h[j];
        }
    }
}

// -----------------------------------------------------------------------------
// Kernel 5: final FC 6 -> 6
// -----------------------------------------------------------------------------
__global__ __launch_bounds__(256) void fc6(
    const float* __restrict__ X, const float* __restrict__ W,
    const float* __restrict__ bias, float* __restrict__ out)
{
    size_t idx = (size_t)blockIdx.x * 256 + threadIdx.x;
    if (idx >= (size_t)BT * 6) return;
    int m = (int)(idx / 6), j = (int)(idx % 6);
    const float* xr = X + (size_t)m * 6;
    float s = bias[j];
    #pragma unroll
    for (int k = 0; k < 6; ++k) s += xr[k] * W[j * 6 + k];
    out[idx] = s;
}

// -----------------------------------------------------------------------------
extern "C" void kernel_launch(void* const* d_in, const int* in_sizes, int n_in,
                              void* d_out, int out_size, void* d_ws, size_t ws_size,
                              hipStream_t stream)
{
    const float* x     = (const float*)d_in[0];
    const float* w1ih0 = (const float*)d_in[1];
    const float* w1hh0 = (const float*)d_in[2];
    const float* b10   = (const float*)d_in[3];
    const float* w1ih1 = (const float*)d_in[4];
    const float* w1hh1 = (const float*)d_in[5];
    const float* b11   = (const float*)d_in[6];
    const float* w2ih0 = (const float*)d_in[7];
    const float* w2hh0 = (const float*)d_in[8];
    const float* b20   = (const float*)d_in[9];
    const float* w2ih1 = (const float*)d_in[10];
    const float* w2hh1 = (const float*)d_in[11];
    const float* b21   = (const float*)d_in[12];
    const float* fcw   = (const float*)d_in[13];
    const float* fcb   = (const float*)d_in[14];
    float* out = (float*)d_out;

    char*  ws  = (char*)d_ws;
    float* xg  = (float*)ws;                          // BT*256 f32  (64 MB)
    float* hA  = (float*)(ws + (size_t)BT * 256 * 4); // BT*64  f32  (16 MB)
    float* hB  = hA + (size_t)BT * 64;                // BT*64  f32  (16 MB)
    float* xgs = hB + (size_t)BT * 64;                // BT*24  f32  (6 MB)
    float* h3  = xgs + (size_t)BT * 24;               // BT*6   f32
    float* h4  = h3 + (size_t)BT * 6;                 // BT*6   f32

    // layer 1.0 (512 -> 64)
    gates256_wmma<<<BT / 128, dim3(256), 0, stream>>>(x, w1ih0, b10, xg, IN_);
    lstm64_wmma  <<<4, dim3(128), 0, stream>>>(xg, w1hh0, hA);
    // layer 1.1 (64 -> 64)
    gates256_wmma<<<BT / 128, dim3(256), 0, stream>>>(hA, w1ih1, b11, xg, H1_);
    lstm64_wmma  <<<4, dim3(128), 0, stream>>>(xg, w1hh1, hB);
    // layer 2.0 (64 -> 6)
    gates_small<<<(BT * 24 + 255) / 256, dim3(256), 0, stream>>>(hB, w2ih0, b20, xgs, H1_);
    lstm6      <<<1, dim3(64), 0, stream>>>(xgs, w2hh0, h3);
    // layer 2.1 (6 -> 6)
    gates_small<<<(BT * 24 + 255) / 256, dim3(256), 0, stream>>>(h3, w2ih1, b21, xgs, H2_);
    lstm6      <<<1, dim3(64), 0, stream>>>(xgs, w2hh1, h4);
    // fc
    fc6<<<(BT * 6 + 255) / 256, dim3(256), 0, stream>>>(h4, fcw, fcb, out);
}
